// CondPredictor_30081950941841
// MI455X (gfx1250) — matrix operand order repaired and checked by reading, not compiled, not run on hardware
//
#include <hip/hip_runtime.h>
#include <cstdint>

#define B_   256
#define T_   128
#define C_   45
#define CP   48
#define NH   512
#define HH   256
#define G4   1024
#define NEGV (-100.0f)

typedef __attribute__((ext_vector_type(16))) __bf16 v16bf;
typedef __attribute__((ext_vector_type(8)))  float  v8f;

union Frag16 { v16bf v; unsigned u[8]; };

__device__ __forceinline__ unsigned short f2bf(float x) {
  unsigned u = __float_as_uint(x);
  u += 0x7fffu + ((u >> 16) & 1u);          // round-to-nearest-even
  return (unsigned short)(u >> 16);
}

__device__ __forceinline__ float sigmoidf(float x) { return 1.0f / (1.0f + expf(-x)); }

// ---- WMMA fragment loaders (CDNA5 16x16x32 bf16 layouts, wave32) ----

// A: 16x32 bf16, element (m,k) at A[m*lda + k].
__device__ __forceinline__ v16bf load_a_frag(const unsigned short* A, int lda) {
  int lane = threadIdx.x & 31;
  int m    = lane & 15;
  int kb   = (lane < 16) ? 0 : 16;
  const unsigned short* row = A + (size_t)m * lda;
  Frag16 f;
#pragma unroll
  for (int v = 0; v < 4; ++v) f.u[v]     = *(const unsigned*)(row + kb + 2 * v);
#pragma unroll
  for (int v = 0; v < 4; ++v) f.u[4 + v] = *(const unsigned*)(row + kb + 8 + 2 * v);
  return f.v;
}

// B: 32x16 bf16 with element (k,n) = W[n*ldw + k]   (W is N x K row-major)
__device__ __forceinline__ v16bf load_b_frag(const unsigned short* W, int ldw) {
  int lane = threadIdx.x & 31;
  int n    = lane & 15;
  int kb   = (lane < 16) ? 0 : 16;
  const unsigned short* row = W + (size_t)n * ldw + kb;
  Frag16 f;
#pragma unroll
  for (int v = 0; v < 8; ++v) f.u[v] = *(const unsigned*)(row + 2 * v);
  return f.v;
}

// B where element (k,n) = X[k*ldx + n] (K along rows in memory)
__device__ __forceinline__ v16bf load_b_frag_kmajor(const unsigned short* X, int ldx) {
  int lane = threadIdx.x & 31;
  int n    = lane & 15;
  int kb   = (lane < 16) ? 0 : 16;
  Frag16 f;
#pragma unroll
  for (int v = 0; v < 8; ++v) {
    unsigned lo = X[(size_t)(kb + 2 * v)     * ldx + n];
    unsigned hi = X[(size_t)(kb + 2 * v + 1) * ldx + n];
    f.u[v] = lo | (hi << 16);
  }
  return f.v;
}

__device__ __forceinline__ v8f wmma_bf16(v16bf a, v16bf b, v8f c) {
  return __builtin_amdgcn_wmma_f32_16x16x32_bf16(false, a, false, b, (short)0, c, false, false);
}

// ---------------- conversion / layout kernels ----------------

__global__ void pad_rows_bf16_kernel(const float* __restrict__ src,
                                     unsigned short* __restrict__ dst,
                                     int N, int K, int Kp) {
  int idx = blockIdx.x * 256 + threadIdx.x;
  if (idx >= N * Kp) return;
  int r = idx / Kp, c = idx % Kp;
  dst[idx] = (c < K) ? f2bf(src[(size_t)r * K + c]) : (unsigned short)0;
}

__global__ void f32_to_bf16_kernel(const float* __restrict__ src,
                                   unsigned short* __restrict__ dst, int n) {
  int idx = blockIdx.x * 256 + threadIdx.x;
  if (idx < n) dst[idx] = f2bf(src[idx]);
}

// cx: (B, CP, 608) bf16 = concat(col_emb[300], db_emb[300], zero-pad)
__global__ void build_cx_kernel(const float* __restrict__ colemb,
                                const float* __restrict__ dbemb,
                                unsigned short* __restrict__ cx) {
  int idx = blockIdx.x * 256 + threadIdx.x;
  if (idx >= B_ * CP * 608) return;
  int k  = idx % 608;
  int rc = idx / 608;
  int c  = rc % CP, b = rc / CP;
  float v = 0.0f;
  if (c < C_ && k < 600)
    v = (k < 300) ? colemb[((size_t)b * C_ + c) * 300 + k]
                  : dbemb[((size_t)b * C_ + c) * 300 + (k - 300)];
  cx[idx] = f2bf(v);
}

// ---------------- generic bf16 GEMM: Out[M,N] = epi(A[M,K] @ W[N,K]^T) ----------------
// block = 256 threads = 8 waves; block tile 32x64; wave tile 16x16
template <int DOBIAS, int DOADD, int DOTANH>
__global__ __launch_bounds__(256) void gemm_bf16_kernel(
    const unsigned short* __restrict__ A, const unsigned short* __restrict__ W,
    const float* __restrict__ bias, const float* __restrict__ addsrc,
    float* __restrict__ Out, int M, int N, int K) {
  __shared__ unsigned short As[32 * 32];
  __shared__ unsigned short Ws[64 * 32];
  const int tid = threadIdx.x;
  const int wv  = tid >> 5;
  const int mi  = wv >> 2;            // 0..1
  const int ni  = wv & 3;             // 0..3
  const int m0  = blockIdx.y * 32;
  const int n0  = blockIdx.x * 64;
  const int ra  = tid >> 3, ca = (tid & 7) * 4;   // A staging coords
  const int rw  = tid >> 2, cw = (tid & 3) * 8;   // W staging coords
  v8f acc = {};
  for (int k0 = 0; k0 < K; k0 += 32) {
    *(uint2*)(As + ra * 32 + ca) =
        *(const uint2*)(A + (size_t)(m0 + ra) * K + k0 + ca);
    *(uint4*)(Ws + rw * 32 + cw) =
        *(const uint4*)(W + (size_t)(n0 + rw) * K + k0 + cw);
    if (k0 + 32 < K) {  // prefetch next K-tile (global_prefetch_b8)
      __builtin_prefetch(A + (size_t)(m0 + ra) * K + k0 + 32 + ca, 0, 1);
      __builtin_prefetch(W + (size_t)(n0 + rw) * K + k0 + 32 + cw, 0, 1);
    }
    __syncthreads();
    v16bf a = load_a_frag(As + mi * 16 * 32, 32);
    v16bf b = load_b_frag(Ws + ni * 16 * 32, 32);
    acc = wmma_bf16(a, b, acc);
    __syncthreads();
  }
  int lane = tid & 31;
  int n    = n0 + ni * 16 + (lane & 15);
  int mb   = m0 + mi * 16 + ((lane < 16) ? 0 : 8);
  float bv = DOBIAS ? bias[n] : 0.0f;
#pragma unroll
  for (int v = 0; v < 8; ++v) {
    size_t o = (size_t)(mb + v) * N + n;
    float x  = acc[v] + bv;
    if (DOADD)  x += addsrc[o];
    if (DOTANH) x = tanhf(x);
    Out[o] = x;
  }
}

// ---------------- LSTM recurrence (one direction, one layer) ----------------
// 16 batch rows / block, 512 threads = 16 waves.
// Wave w owns hidden slice [w*16, w*16+16): N-tiles {w, w+16, w+32, w+48} = i,f,g,o
// for the same hidden units -> gates never leave the wave, c stays in registers.
// Only h (16x256 bf16, 8KB) lives in LDS. K-loop kept rolled so Wh streams from
// L2 each step (512KB, fully L2-resident) instead of being hoisted + spilled.
__global__ __launch_bounds__(512) void lstm_step_kernel(
    const float* __restrict__ xw,           // (B, TS, 1024) f32
    const unsigned short* __restrict__ Wh,  // (1024, 256) bf16
    const int* __restrict__ lens,
    float* __restrict__ out,                // (B, TS, 512) f32, write at colOff
    int Tlen, int TS, int colOff, int dirBwd) {
  __shared__ unsigned short h_bf[16 * HH];  // 8 KB
  const int tid  = threadIdx.x;
  const int w    = tid >> 5;                // wave 0..15 -> hidden slice
  const int lane = tid & 31;
  const int b0   = blockIdx.x * 16;
  const int nn   = lane & 15;
  const int mb   = (lane < 16) ? 0 : 8;
  const int hh   = w * 16 + nn;             // hidden unit this lane owns

  for (int i = tid; i < 16 * HH; i += 512) h_bf[i] = 0;
  int lenv[8];
#pragma unroll
  for (int v = 0; v < 8; ++v) lenv[v] = lens[b0 + mb + v];
  float c_reg[8];
#pragma unroll
  for (int v = 0; v < 8; ++v) c_reg[v] = 0.0f;
  // prefetch tiling: thread -> (row w, 128B chunk lane) covers the block's
  // 16 x 4KB next-step xw slice with one global_prefetch_b8 per thread
  const int   pfrow = b0 + w;
  const int   pflen = lens[pfrow];
  const int   pfoff = lane * 32;            // 32 floats = 128B line
  __syncthreads();

  for (int s = 0; s < Tlen; ++s) {
    v8f acc[4] = {{}, {}, {}, {}};
#pragma unroll 1
    for (int k0 = 0; k0 < HH; k0 += 32) {
      v16bf a = load_a_frag(h_bf + k0, HH);
#pragma unroll
      for (int g = 0; g < 4; ++g) {
        // gate g, hidden slice w: weight rows [g*256 + w*16, +16)
        v16bf b = load_b_frag(Wh + (size_t)(g * 256 + w * 16) * HH + k0, HH);
        acc[g] = wmma_bf16(a, b, acc[g]);
      }
    }
    // prefetch next step's xw rows while this step's gate math runs
    {
      int sn = s + 1;
      if (sn < pflen) {
        int p = dirBwd ? (pflen - 1 - sn) : sn;
        __builtin_prefetch(xw + ((size_t)pfrow * TS + p) * G4 + pfoff, 0, 1);
      }
    }
    __syncthreads();  // all waves done reading h_bf for this step
#pragma unroll
    for (int v = 0; v < 8; ++v) {
      int m   = mb + v;
      int b   = b0 + m;
      int len = lenv[v];
      if (s < len) {
        int p = dirBwd ? (len - 1 - s) : s;
        const float* xr = xw + ((size_t)b * TS + p) * G4;
        float gi = acc[0][v] + xr[hh];
        float gf = acc[1][v] + xr[hh + 256];
        float gg = acc[2][v] + xr[hh + 512];
        float go = acc[3][v] + xr[hh + 768];
        float cn = sigmoidf(gf) * c_reg[v] + sigmoidf(gi) * tanhf(gg);
        float hn = sigmoidf(go) * tanhf(cn);
        c_reg[v] = cn;
        h_bf[m * HH + hh] = f2bf(hn);
        out[(size_t)b * TS * NH + (size_t)p * NH + colOff + hh] = hn;
      }
    }
    __syncthreads();  // h updated before next step's A-fragment reads
  }
}

// ---------------- batched attention scores: att[b,c,t] = col_enc[b,c,:].qa[b,t,:] ----
__global__ __launch_bounds__(256) void attn_score_kernel(
    const unsigned short* __restrict__ colenc,  // (B,CP,512) bf16
    const unsigned short* __restrict__ qa,      // (B,T,512)  bf16
    const int* __restrict__ qlen, const int* __restrict__ clen,  // clen may be null
    float* __restrict__ att) {                  // (B,CP,T)
  int b    = blockIdx.y;
  int wv   = threadIdx.x >> 5;
  int tile = blockIdx.x * 8 + wv;               // 0..23
  int ct   = tile >> 3;                         // 0..2
  int tt   = tile & 7;                          // 0..7
  const unsigned short* Ab = colenc + ((size_t)b * CP + ct * 16) * NH;
  const unsigned short* Bb = qa + ((size_t)b * T_ + tt * 16) * NH;
  v8f acc = {};
#pragma unroll 1
  for (int k0 = 0; k0 < NH; k0 += 32) {
    v16bf a = load_a_frag(Ab + k0, NH);
    v16bf bb = load_b_frag(Bb + k0, NH);
    acc = wmma_bf16(a, bb, acc);
  }
  int lane = threadIdx.x & 31;
  int t  = tt * 16 + (lane & 15);
  int c0 = ct * 16 + ((lane < 16) ? 0 : 8);
  int ql = qlen[b];
  int cl = clen ? clen[b] : CP;
#pragma unroll
  for (int v = 0; v < 8; ++v) {
    int c = c0 + v;
    float x = acc[v];
    if (t >= ql) x = NEGV;
    if (c >= cl || c >= C_) x = NEGV;
    att[((size_t)b * CP + c) * T_ + t] = x;
  }
}

// ---------------- wave softmax over rows of length 128 (in place) ----------------
__global__ __launch_bounds__(256) void softmax128_kernel(
    float* __restrict__ X, unsigned short* __restrict__ Pbf, int R) {
  int row = blockIdx.x * 8 + (threadIdx.x >> 5);
  if (row >= R) return;
  int lane = threadIdx.x & 31;
  float* x = X + (size_t)row * T_;
  float v[4];
  float mx = -3.4e38f;
#pragma unroll
  for (int i = 0; i < 4; ++i) { v[i] = x[lane * 4 + i]; mx = fmaxf(mx, v[i]); }
  for (int off = 16; off; off >>= 1) mx = fmaxf(mx, __shfl_xor(mx, off, 32));
  float sum = 0.0f;
#pragma unroll
  for (int i = 0; i < 4; ++i) { v[i] = expf(v[i] - mx); sum += v[i]; }
  for (int off = 16; off; off >>= 1) sum += __shfl_xor(sum, off, 32);
  float inv = 1.0f / sum;
#pragma unroll
  for (int i = 0; i < 4; ++i) {
    float p = v[i] * inv;
    x[lane * 4 + i] = p;
    if (Pbf) Pbf[(size_t)row * T_ + lane * 4 + i] = f2bf(p);
  }
}

// s[b,t] = sum_{c<45} p[b,c,t]
__global__ void colsum_kernel(const float* __restrict__ p, float* __restrict__ s) {
  int idx = blockIdx.x * 256 + threadIdx.x;
  if (idx >= B_ * T_) return;
  int t = idx % T_, b = idx / T_;
  float acc = 0.0f;
  for (int c = 0; c < C_; ++c) acc += p[((size_t)b * CP + c) * T_ + t];
  s[idx] = acc;
}

// q_w_num[b,h] = sum_t s[b,t]*q_enc[b,t,h]
__global__ void qwnum_kernel(const float* __restrict__ s, const float* __restrict__ qenc,
                             float* __restrict__ out) {
  int idx = blockIdx.x * 256 + threadIdx.x;
  if (idx >= B_ * NH) return;
  int h = idx % NH, b = idx / NH;
  float acc = 0.0f;
  for (int t = 0; t < T_; ++t)
    acc += s[b * T_ + t] * qenc[((size_t)b * T_ + t) * NH + h];
  out[idx] = acc;
}

// q_w[b,c,h] = sum_t p_qc[b,c,t] * q_enc[b,t,h]   (batched WMMA, K=128)
__global__ __launch_bounds__(256) void qw_gemm_kernel(
    const unsigned short* __restrict__ Pbf,   // (B,CP,128) bf16
    const unsigned short* __restrict__ qenc,  // (B,T,512)  bf16
    float* __restrict__ qw) {                 // (B,CP,512)
  int b    = blockIdx.y;
  int wv   = threadIdx.x >> 5;
  int tile = blockIdx.x * 8 + wv;             // 0..95
  int ct   = tile >> 5;                       // 0..2
  int ht   = tile & 31;                       // 0..31
  const unsigned short* Ab = Pbf + ((size_t)b * CP + ct * 16) * T_;
  const unsigned short* Bb = qenc + (size_t)b * T_ * NH + ht * 16;
  v8f acc = {};
#pragma unroll 1
  for (int k0 = 0; k0 < T_; k0 += 32) {
    v16bf a  = load_a_frag(Ab + k0, T_);
    v16bf bb = load_b_frag_kmajor(Bb + (size_t)k0 * NH, NH);
    acc = wmma_bf16(a, bb, acc);
  }
  int lane = threadIdx.x & 31;
  int h  = ht * 16 + (lane & 15);
  int c0 = ct * 16 + ((lane < 16) ? 0 : 8);
#pragma unroll
  for (int v = 0; v < 8; ++v)
    qw[((size_t)b * CP + c0 + v) * NH + h] = acc[v];
}

// col_sel[b,k,:] = col_enc[b, gt[b,k], :]
__global__ void colsel_kernel(const float* __restrict__ colenc, const int* __restrict__ gt,
                              float* __restrict__ sel, unsigned short* __restrict__ selbf) {
  int idx = blockIdx.x * 256 + threadIdx.x;
  if (idx >= B_ * 5 * NH) return;
  int h = idx % NH;
  int rk = idx / NH;
  int k = rk % 5, b = rk / 5;
  int g = gt[b * 5 + k];
  float v = colenc[((size_t)b * CP + g) * NH + h];
  sel[idx] = v;
  selbf[idx] = f2bf(v);
}

// att3[b,k,t] = oa[b,t,:].col_sel[b,k,:]   (masked)
__global__ void att3_kernel(const float* __restrict__ oa, const float* __restrict__ sel,
                            const int* __restrict__ qlen, float* __restrict__ att3) {
  int idx = blockIdx.x * 256 + threadIdx.x;
  if (idx >= B_ * 5 * T_) return;
  int t = idx % T_;
  int rk = idx / T_;
  int k = rk % 5, b = rk / 5;
  const float* pa = oa + ((size_t)b * T_ + t) * NH;
  const float* pc = sel + (size_t)(b * 5 + k) * NH;
  float s = 0.0f;
  for (int h = 0; h < NH; ++h) s += pa[h] * pc[h];
  att3[idx] = (t < qlen[b]) ? s : NEGV;
}

// q_w_op[b,k,h] = sum_t p_op[b,k,t]*q_enc[b,t,h]
__global__ void qwop_kernel(const float* __restrict__ pop, const float* __restrict__ qenc,
                            float* __restrict__ out) {
  int idx = blockIdx.x * 256 + threadIdx.x;
  if (idx >= B_ * 5 * NH) return;
  int h = idx % NH;
  int rk = idx / NH;
  int k = rk % 5, b = rk / 5;
  float s = 0.0f;
  for (int t = 0; t < T_; ++t)
    s += pop[(size_t)(b * 5 + k) * T_ + t] * qenc[((size_t)b * T_ + t) * NH + h];
  out[idx] = s;
}

// out[r,o] = g[r,:].W[o,:] + bias[o]
__global__ void dot_head_kernel(const float* __restrict__ g, const float* __restrict__ W,
                                const float* __restrict__ bias, float* __restrict__ out,
                                int R, int O) {
  int idx = blockIdx.x * 256 + threadIdx.x;
  if (idx >= R * O) return;
  int r = idx / O, o = idx % O;
  const float* gr = g + (size_t)r * NH;
  const float* wr = W + (size_t)o * NH;
  float s = bias[o];
  for (int h = 0; h < NH; ++h) s += gr[h] * wr[h];
  out[idx] = s;
}

// col_score[b,c] = mask(g[b,c,:].W + b0)
__global__ void col_head_kernel(const float* __restrict__ g, const float* __restrict__ W,
                                const float* __restrict__ bias, const int* __restrict__ clen,
                                float* __restrict__ out) {
  int idx = blockIdx.x * 256 + threadIdx.x;
  if (idx >= B_ * C_) return;
  int b = idx / C_, c = idx % C_;
  const float* gr = g + ((size_t)b * CP + c) * NH;
  float s = bias[0];
  for (int h = 0; h < NH; ++h) s += gr[h] * W[h];
  out[idx] = (c < clen[b]) ? s : NEGV;
}

// ---------------- host ----------------

static inline int cdiv(int a, int b) { return (a + b - 1) / b; }

static void launch_gemm(int epi, const unsigned short* A, const unsigned short* W,
                        const float* bias, const float* add, float* Out,
                        int M, int N, int K, hipStream_t st) {
  dim3 g(N / 64, M / 32);
  if (epi == 0)
    gemm_bf16_kernel<1, 0, 0><<<g, 256, 0, st>>>(A, W, bias, nullptr, Out, M, N, K);
  else if (epi == 1)
    gemm_bf16_kernel<1, 0, 1><<<g, 256, 0, st>>>(A, W, bias, nullptr, Out, M, N, K);
  else
    gemm_bf16_kernel<1, 1, 1><<<g, 256, 0, st>>>(A, W, bias, add, Out, M, N, K);
}

extern "C" void kernel_launch(void* const* d_in, const int* in_sizes, int n_in,
                              void* d_out, int out_size, void* d_ws, size_t ws_size,
                              hipStream_t stream) {
  const float* q_emb   = (const float*)d_in[0];
  const int*   q_len   = (const int*)d_in[1];
  const float* col_emb = (const float*)d_in[2];
  const int*   col_len = (const int*)d_in[3];
  const float* db_emb  = (const float*)d_in[4];
  const int*   gt      = (const int*)d_in[5];
  // params flattened as jax pytree: dict keys sorted alphabetically
  // 6:col_lstm(12) 18:col_num_out 20:col_num_out_q 22:col_out 24:col_out_c
  // 26:col_out_q 28:op_att 30:op_out 32:op_out_c 34:op_out_q 36:q_att
  // 38:q_lstm(12) 50:q_num_att
  auto LW = [&](int base, int layer, int dir, int which) -> const float* {
    return (const float*)d_in[base + layer * 6 + dir * 3 + which];
  };

  size_t off = 0;
  auto alloc = [&](size_t bytes) -> char* {
    off = (off + 255) & ~(size_t)255;
    char* p = (char*)d_ws + off;
    off += bytes;
    return p;
  };

  // bf16 weight buffers
  unsigned short *w_q_wi0[2], *w_q_wh[2][2], *w_q_wi1[2];
  unsigned short *w_c_wi0[2], *w_c_wh[2][2], *w_c_wi1[2];
  for (int d = 0; d < 2; ++d) {
    w_q_wi0[d]   = (unsigned short*)alloc((size_t)G4 * 320 * 2);
    w_q_wh[0][d] = (unsigned short*)alloc((size_t)G4 * HH * 2);
    w_q_wi1[d]   = (unsigned short*)alloc((size_t)G4 * NH * 2);
    w_q_wh[1][d] = (unsigned short*)alloc((size_t)G4 * HH * 2);
    w_c_wi0[d]   = (unsigned short*)alloc((size_t)G4 * 608 * 2);
    w_c_wh[0][d] = (unsigned short*)alloc((size_t)G4 * HH * 2);
    w_c_wi1[d]   = (unsigned short*)alloc((size_t)G4 * NH * 2);
    w_c_wh[1][d] = (unsigned short*)alloc((size_t)G4 * HH * 2);
  }
  unsigned short* w_qnumatt = (unsigned short*)alloc((size_t)NH * NH * 2);
  unsigned short* w_qatt    = (unsigned short*)alloc((size_t)NH * NH * 2);
  unsigned short* w_opatt   = (unsigned short*)alloc((size_t)NH * NH * 2);
  unsigned short* w_cnoq    = (unsigned short*)alloc((size_t)NH * NH * 2);
  unsigned short* w_coq     = (unsigned short*)alloc((size_t)NH * NH * 2);
  unsigned short* w_coc     = (unsigned short*)alloc((size_t)NH * NH * 2);
  unsigned short* w_ooq     = (unsigned short*)alloc((size_t)NH * NH * 2);
  unsigned short* w_ooc     = (unsigned short*)alloc((size_t)NH * NH * 2);

  const int MQ = B_ * T_;   // 32768
  const int MC = B_ * CP;   // 12288
  unsigned short* qx = (unsigned short*)alloc((size_t)MQ * 320 * 2);
  unsigned short* cx = (unsigned short*)alloc((size_t)MC * 608 * 2);
  float* xw = (float*)alloc((size_t)MQ * G4 * 4);   // reused region for phase-3 temps
  float* q_cat0 = (float*)alloc((size_t)MQ * NH * 4);
  unsigned short* q_cat0_bf = (unsigned short*)alloc((size_t)MQ * NH * 2);
  float* q_enc = (float*)alloc((size_t)MQ * NH * 4);
  unsigned short* q_enc_bf = (unsigned short*)alloc((size_t)MQ * NH * 2);
  float* c_cat0 = (float*)alloc((size_t)MC * NH * 4);
  unsigned short* c_cat0_bf = (unsigned short*)alloc((size_t)MC * NH * 2);
  float* col_enc = (float*)alloc((size_t)MC * NH * 4);
  unsigned short* col_enc_bf = (unsigned short*)alloc((size_t)MC * NH * 2);

  // proj aliases q_cat0 (free after layer-1 projections)
  float* proj = q_cat0;
  unsigned short* proj_bf = q_cat0_bf;

  // phase-3 temporaries carved out of the xw region (free after LSTMs)
  size_t off2 = 0;
  auto alloc2 = [&](size_t bytes) -> char* {
    off2 = (off2 + 255) & ~(size_t)255;
    char* p = (char*)xw + off2;
    off2 += bytes;
    return p;
  };
  float* att            = (float*)alloc2((size_t)MC * T_ * 4);
  unsigned short* pqcbf = (unsigned short*)alloc2((size_t)MC * T_ * 2);
  float* sbuf           = (float*)alloc2((size_t)B_ * T_ * 4);
  float* q_w_num        = (float*)alloc2((size_t)B_ * NH * 4);
  unsigned short* q_w_num_bf = (unsigned short*)alloc2((size_t)B_ * NH * 2);
  float* t1             = (float*)alloc2((size_t)B_ * NH * 4);
  float* q_w            = (float*)alloc2((size_t)MC * NH * 4);
  unsigned short* q_w_bf = (unsigned short*)alloc2((size_t)MC * NH * 2);
  float* tmp_cs         = (float*)alloc2((size_t)MC * NH * 4);
  float* col_sel        = (float*)alloc2((size_t)B_ * 5 * NH * 4);
  unsigned short* col_sel_bf = (unsigned short*)alloc2((size_t)B_ * 5 * NH * 2);
  float* att3           = (float*)alloc2((size_t)B_ * 5 * T_ * 4);
  float* q_w_op         = (float*)alloc2((size_t)B_ * 5 * NH * 4);
  unsigned short* q_w_op_bf = (unsigned short*)alloc2((size_t)B_ * 5 * NH * 2);
  float* tmp_op         = (float*)alloc2((size_t)B_ * 5 * NH * 4);

  float* out_colnum = (float*)d_out;            // (256,6)
  float* out_col    = (float*)d_out + 1536;     // (256,45)
  float* out_op     = (float*)d_out + 13056;    // (256,5,12)

  // ---- Phase 0: weight/input conversion + zero init ----
  for (int d = 0; d < 2; ++d) {
    pad_rows_bf16_kernel<<<cdiv(G4 * 320, 256), 256, 0, stream>>>(LW(38, 0, d, 0), w_q_wi0[d], G4, 300, 320);
    pad_rows_bf16_kernel<<<cdiv(G4 * HH, 256), 256, 0, stream>>>(LW(38, 0, d, 1), w_q_wh[0][d], G4, HH, HH);
    pad_rows_bf16_kernel<<<cdiv(G4 * NH, 256), 256, 0, stream>>>(LW(38, 1, d, 0), w_q_wi1[d], G4, NH, NH);
    pad_rows_bf16_kernel<<<cdiv(G4 * HH, 256), 256, 0, stream>>>(LW(38, 1, d, 1), w_q_wh[1][d], G4, HH, HH);
    pad_rows_bf16_kernel<<<cdiv(G4 * 608, 256), 256, 0, stream>>>(LW(6, 0, d, 0), w_c_wi0[d], G4, 600, 608);
    pad_rows_bf16_kernel<<<cdiv(G4 * HH, 256), 256, 0, stream>>>(LW(6, 0, d, 1), w_c_wh[0][d], G4, HH, HH);
    pad_rows_bf16_kernel<<<cdiv(G4 * NH, 256), 256, 0, stream>>>(LW(6, 1, d, 0), w_c_wi1[d], G4, NH, NH);
    pad_rows_bf16_kernel<<<cdiv(G4 * HH, 256), 256, 0, stream>>>(LW(6, 1, d, 1), w_c_wh[1][d], G4, HH, HH);
  }
  const int NLIN = NH * NH;
  pad_rows_bf16_kernel<<<cdiv(NLIN, 256), 256, 0, stream>>>((const float*)d_in[50], w_qnumatt, NH, NH, NH);
  pad_rows_bf16_kernel<<<cdiv(NLIN, 256), 256, 0, stream>>>((const float*)d_in[36], w_qatt, NH, NH, NH);
  pad_rows_bf16_kernel<<<cdiv(NLIN, 256), 256, 0, stream>>>((const float*)d_in[28], w_opatt, NH, NH, NH);
  pad_rows_bf16_kernel<<<cdiv(NLIN, 256), 256, 0, stream>>>((const float*)d_in[20], w_cnoq, NH, NH, NH);
  pad_rows_bf16_kernel<<<cdiv(NLIN, 256), 256, 0, stream>>>((const float*)d_in[26], w_coq, NH, NH, NH);
  pad_rows_bf16_kernel<<<cdiv(NLIN, 256), 256, 0, stream>>>((const float*)d_in[24], w_coc, NH, NH, NH);
  pad_rows_bf16_kernel<<<cdiv(NLIN, 256), 256, 0, stream>>>((const float*)d_in[34], w_ooq, NH, NH, NH);
  pad_rows_bf16_kernel<<<cdiv(NLIN, 256), 256, 0, stream>>>((const float*)d_in[32], w_ooc, NH, NH, NH);
  pad_rows_bf16_kernel<<<cdiv(MQ * 320, 256), 256, 0, stream>>>(q_emb, qx, MQ, 300, 320);
  build_cx_kernel<<<cdiv(B_ * CP * 608, 256), 256, 0, stream>>>(col_emb, db_emb, cx);
  (void)hipMemsetAsync(q_cat0, 0, (size_t)MQ * NH * 4, stream);
  (void)hipMemsetAsync(q_enc, 0, (size_t)MQ * NH * 4, stream);
  (void)hipMemsetAsync(c_cat0, 0, (size_t)MC * NH * 4, stream);
  (void)hipMemsetAsync(col_enc, 0, (size_t)MC * NH * 4, stream);

  // ---- Phase 1: question BiLSTM ----
  for (int d = 0; d < 2; ++d) {
    launch_gemm(0, qx, w_q_wi0[d], LW(38, 0, d, 2), nullptr, xw, MQ, G4, 320, stream);
    lstm_step_kernel<<<16, 512, 0, stream>>>(xw, w_q_wh[0][d], q_len, q_cat0, T_, T_, d ? HH : 0, d);
  }
  f32_to_bf16_kernel<<<cdiv(MQ * NH, 256), 256, 0, stream>>>(q_cat0, q_cat0_bf, MQ * NH);
  for (int d = 0; d < 2; ++d) {
    launch_gemm(0, q_cat0_bf, w_q_wi1[d], LW(38, 1, d, 2), nullptr, xw, MQ, G4, NH, stream);
    lstm_step_kernel<<<16, 512, 0, stream>>>(xw, w_q_wh[1][d], q_len, q_enc, T_, T_, d ? HH : 0, d);
  }
  f32_to_bf16_kernel<<<cdiv(MQ * NH, 256), 256, 0, stream>>>(q_enc, q_enc_bf, MQ * NH);

  // ---- Phase 2: column BiLSTM ----
  for (int d = 0; d < 2; ++d) {
    launch_gemm(0, cx, w_c_wi0[d], LW(6, 0, d, 2), nullptr, xw, MC, G4, 608, stream);
    lstm_step_kernel<<<16, 512, 0, stream>>>(xw, w_c_wh[0][d], col_len, c_cat0, C_, CP, d ? HH : 0, d);
  }
  f32_to_bf16_kernel<<<cdiv(MC * NH, 256), 256, 0, stream>>>(c_cat0, c_cat0_bf, MC * NH);
  for (int d = 0; d < 2; ++d) {
    launch_gemm(0, c_cat0_bf, w_c_wi1[d], LW(6, 1, d, 2), nullptr, xw, MC, G4, NH, stream);
    lstm_step_kernel<<<16, 512, 0, stream>>>(xw, w_c_wh[1][d], col_len, col_enc, C_, CP, d ? HH : 0, d);
  }
  f32_to_bf16_kernel<<<cdiv(MC * NH, 256), 256, 0, stream>>>(col_enc, col_enc_bf, MC * NH);

  // ---- Phase 3a: col-num head ----
  launch_gemm(0, q_enc_bf, w_qnumatt, (const float*)d_in[51], nullptr, proj, MQ, NH, NH, stream);
  f32_to_bf16_kernel<<<cdiv(MQ * NH, 256), 256, 0, stream>>>(proj, proj_bf, MQ * NH);
  attn_score_kernel<<<dim3(3, B_), 256, 0, stream>>>(col_enc_bf, proj_bf, q_len, col_len, att);
  softmax128_kernel<<<MC / 8, 256, 0, stream>>>(att, nullptr, MC);
  colsum_kernel<<<cdiv(B_ * T_, 256), 256, 0, stream>>>(att, sbuf);
  qwnum_kernel<<<cdiv(B_ * NH, 256), 256, 0, stream>>>(sbuf, q_enc, q_w_num);
  f32_to_bf16_kernel<<<cdiv(B_ * NH, 256), 256, 0, stream>>>(q_w_num, q_w_num_bf, B_ * NH);
  launch_gemm(1, q_w_num_bf, w_cnoq, (const float*)d_in[21], nullptr, t1, B_, NH, NH, stream);
  dot_head_kernel<<<cdiv(B_ * 6, 256), 256, 0, stream>>>(t1, (const float*)d_in[18], (const float*)d_in[19], out_colnum, B_, 6);

  // ---- Phase 3b: col head ----
  launch_gemm(0, q_enc_bf, w_qatt, (const float*)d_in[37], nullptr, proj, MQ, NH, NH, stream);
  f32_to_bf16_kernel<<<cdiv(MQ * NH, 256), 256, 0, stream>>>(proj, proj_bf, MQ * NH);
  attn_score_kernel<<<dim3(3, B_), 256, 0, stream>>>(col_enc_bf, proj_bf, q_len, nullptr, att);
  softmax128_kernel<<<MC / 8, 256, 0, stream>>>(att, pqcbf, MC);
  qw_gemm_kernel<<<dim3(12, B_), 256, 0, stream>>>(pqcbf, q_enc_bf, q_w);
  f32_to_bf16_kernel<<<cdiv(MC * NH, 256), 256, 0, stream>>>(q_w, q_w_bf, MC * NH);
  launch_gemm(0, q_w_bf, w_coq, (const float*)d_in[27], nullptr, tmp_cs, MC, NH, NH, stream);
  launch_gemm(2, col_enc_bf, w_coc, (const float*)d_in[25], tmp_cs, q_w, MC, NH, NH, stream);  // g -> q_w
  col_head_kernel<<<cdiv(B_ * C_, 256), 256, 0, stream>>>(q_w, (const float*)d_in[22], (const float*)d_in[23], col_len, out_col);

  // ---- Phase 3c: op head ----
  colsel_kernel<<<cdiv(B_ * 5 * NH, 256), 256, 0, stream>>>(col_enc, gt, col_sel, col_sel_bf);
  launch_gemm(0, q_enc_bf, w_opatt, (const float*)d_in[29], nullptr, proj, MQ, NH, NH, stream);
  att3_kernel<<<cdiv(B_ * 5 * T_, 256), 256, 0, stream>>>(proj, col_sel, q_len, att3);
  softmax128_kernel<<<(B_ * 5) / 8, 256, 0, stream>>>(att3, nullptr, B_ * 5);
  qwop_kernel<<<cdiv(B_ * 5 * NH, 256), 256, 0, stream>>>(att3, q_enc, q_w_op);
  f32_to_bf16_kernel<<<cdiv(B_ * 5 * NH, 256), 256, 0, stream>>>(q_w_op, q_w_op_bf, B_ * 5 * NH);
  launch_gemm(0, q_w_op_bf, w_ooq, (const float*)d_in[35], nullptr, tmp_op, B_ * 5, NH, NH, stream);
  launch_gemm(2, col_sel_bf, w_ooc, (const float*)d_in[33], tmp_op, q_w_op, B_ * 5, NH, NH, stream);  // g2 -> q_w_op
  dot_head_kernel<<<cdiv(B_ * 5 * 12, 256), 256, 0, stream>>>(q_w_op, (const float*)d_in[30], (const float*)d_in[31], out_op, B_ * 5, 12);
}
